// MultiHeadAttention_45715631899267
// MI455X (gfx1250) — matrix operand order repaired and checked by reading, not compiled, not run on hardware
//
#include <hip/hip_runtime.h>
#include <hip/hip_bf16.h>

typedef __bf16 bf16;
typedef __attribute__((ext_vector_type(16))) __bf16 v16bf;
typedef __attribute__((ext_vector_type(8)))  float  v8f;

#define WMMA_BF16(a, b, c) \
  __builtin_amdgcn_wmma_f32_16x16x32_bf16(false, (a), false, (b), (short)0, (c), false, false)

// Pin memory ops issued before this point: prevents the machine scheduler
// from sinking the fragment loads down next to their consuming WMMAs.
#define SCHED_FENCE() __builtin_amdgcn_sched_barrier(0)

// ---------------- geometry ----------------
constexpr int BATCH = 2;
constexpr int SEQ   = 2048;
constexpr int DMODEL= 1024;
constexpr int NHEAD = 16;
constexpr int HD    = 64;              // head dim
constexpr int MROWS = BATCH * SEQ;     // 4096 flattened rows

// ---------------- fragment loaders ----------------
// A fragment (16x32, 16-bit): lane m = lane&15, kg = lane>>4.
//   elem i (0..15) -> K = kg*16 + i
__device__ __forceinline__ v16bf load_a_frag(const bf16* __restrict__ base, int ld,
                                             int row_base, int kbase, int lane) {
  const int lm = lane & 15, kg = lane >> 4;
  const bf16* p = base + (size_t)(row_base + lm) * ld + kbase + kg * 16;
  v16bf a;
#pragma unroll
  for (int i = 0; i < 16; ++i) a[i] = p[i];
  return a;
}

// B fragment (32x16, 16-bit), source column-major (row n holds the K run):
//   lane n = lane&15, elem i -> K = kg*16 + i  (16 contiguous elements)
__device__ __forceinline__ v16bf load_b_frag(const bf16* __restrict__ base, int ld,
                                             int n_base, int kbase, int lane) {
  const int lm = lane & 15, kg = lane >> 4;
  const bf16* p = base + (size_t)(n_base + lm) * ld + kbase + kg * 16;
  v16bf b;
#pragma unroll
  for (int i = 0; i < 16; ++i) b[i] = p[i];
  return b;
}

// Same B pattern but reading a chunk-local LDS K-stage (32 keys x 64 dims, linear)
__device__ __forceinline__ v16bf load_b_frag_lds(const bf16* lds, int key_base,
                                                 int kbase, int lane) {
  const int lm = lane & 15, kg = lane >> 4;
  const bf16* p = lds + (key_base + lm) * HD + kbase + kg * 16;
  v16bf b;
#pragma unroll
  for (int i = 0; i < 16; ++i) b[i] = p[i];
  return b;
}

// Async copy of one 4 KiB K chunk (32 keys x 64 bf16) into LDS: 8 x b128, 16 B/lane.
// ASYNCcnt += 8 ; completes in order.
__device__ __forceinline__ void async_copy_chunk(const bf16* gsrc, const bf16* lds_dst,
                                                 int lane) {
  const unsigned long long g0 = (unsigned long long)(const char*)gsrc + (unsigned)(lane * 16);
  const unsigned l0 = (unsigned)(unsigned long long)lds_dst + (unsigned)(lane * 16);
#pragma unroll
  for (int i = 0; i < 8; ++i) {
    const unsigned long long ga = g0 + (unsigned)(i * 512);
    const unsigned la = l0 + (unsigned)(i * 512);
    asm volatile("global_load_async_to_lds_b128 %0, %1, off"
                 :: "v"(la), "v"(ga) : "memory");
  }
}

// ---------------- kernel 0: fp32 -> bf16 cast ----------------
__global__ void cast_bf16_kernel(const float* __restrict__ src, bf16* __restrict__ dst, int n) {
  int i = blockIdx.x * blockDim.x + threadIdx.x;
  if (i < n) dst[i] = (bf16)src[i];
}

// Wt[n*1024 + k] = W[k*1024 + n]  (column-major copy for B fragments)
__global__ void transpose_cast_kernel(const float* __restrict__ W, bf16* __restrict__ Wt) {
  int i = blockIdx.x * blockDim.x + threadIdx.x;
  int n = i >> 10, k = i & 1023;
  Wt[i] = (bf16)W[(size_t)k * DMODEL + n];
}

// ---------------- kernel 1: fused QKV GEMM + RoPE ----------------
// One wave -> 16x64 tile (one head's columns) of Q, K, V simultaneously.
__global__ void __launch_bounds__(32)
qkv_rope_kernel(const bf16* __restrict__ xb,
                const bf16* __restrict__ Wqt, const bf16* __restrict__ Wkt,
                const bf16* __restrict__ Wvt,
                bf16* __restrict__ Qb, bf16* __restrict__ Kb, bf16* __restrict__ Vt) {
  const int lane = threadIdx.x & 31;
  const int lm = lane & 15, kg = lane >> 4;
  const int mtile = blockIdx.x;          // 0..255
  const int h     = blockIdx.y;          // 0..15 == head
  const int row_base = mtile * 16;
  const int nb = h * HD;                 // column base in d_model

  v8f accQ[4] = {}, accK[4] = {}, accV[4] = {};

#pragma unroll 2
  for (int kb = 0; kb < DMODEL; kb += 32) {
    __builtin_prefetch(xb + (size_t)(row_base + lm) * DMODEL + kb + 64, 0, 1);
    // issue ALL 13 fragments of this k-step, then fence the scheduler so the
    // 26 b128 loads stay batched ahead of the WMMA block (latencies overlap)
    const v16bf a = load_a_frag(xb, DMODEL, row_base, kb, lane);
    v16bf bq[4], bk[4], bv[4];
#pragma unroll
    for (int f = 0; f < 4; ++f) {
      bq[f] = load_b_frag(Wqt, DMODEL, nb + f * 16, kb, lane);
      bk[f] = load_b_frag(Wkt, DMODEL, nb + f * 16, kb, lane);
      bv[f] = load_b_frag(Wvt, DMODEL, nb + f * 16, kb, lane);
    }
    SCHED_FENCE();
#pragma unroll
    for (int f = 0; f < 4; ++f) {
      accQ[f] = WMMA_BF16(a, bq[f], accQ[f]);
      accK[f] = WMMA_BF16(a, bk[f], accK[f]);
      accV[f] = WMMA_BF16(a, bv[f], accV[f]);
    }
  }

  // RoPE: pair (n, n+32) within the head -> fragments (0,2) and (1,3).
  const float LOG_THETA = 9.210340371976184f;             // ln(10000)
  const float invf0 = __expf(-(float)(2 * lm)        * (1.0f / 64.0f) * LOG_THETA);
  const float invf1 = __expf(-(float)(2 * (16 + lm)) * (1.0f / 64.0f) * LOG_THETA);
  const float qscale = 0.125f;                            // 1/sqrt(hd)

#pragma unroll
  for (int vr = 0; vr < 8; ++vr) {
    const int row  = row_base + vr + 8 * kg;              // D-frag: M = vr + 8*(lane>=16)
    const int b    = row >> 11;
    const int sidx = row & (SEQ - 1);
    float c0, s0, c1, s1;
    __sincosf((float)sidx * invf0, &s0, &c0);
    __sincosf((float)sidx * invf1, &s1, &c1);

    const float q0 = accQ[0][vr], q1 = accQ[1][vr], q2 = accQ[2][vr], q3 = accQ[3][vr];
    const float k0 = accK[0][vr], k1 = accK[1][vr], k2 = accK[2][vr], k3 = accK[3][vr];

    const size_t qkbase = (((size_t)(b * NHEAD + h)) * SEQ + sidx) * HD;
    Qb[qkbase +  0 + lm] = (bf16)((q0 * c0 - q2 * s0) * qscale);
    Qb[qkbase + 16 + lm] = (bf16)((q1 * c1 - q3 * s1) * qscale);
    Qb[qkbase + 32 + lm] = (bf16)((q2 * c0 + q0 * s0) * qscale);
    Qb[qkbase + 48 + lm] = (bf16)((q3 * c1 + q1 * s1) * qscale);

    Kb[qkbase +  0 + lm] = (bf16)(k0 * c0 - k2 * s0);
    Kb[qkbase + 16 + lm] = (bf16)(k1 * c1 - k3 * s1);
    Kb[qkbase + 32 + lm] = (bf16)(k2 * c0 + k0 * s0);
    Kb[qkbase + 48 + lm] = (bf16)(k3 * c1 + k1 * s1);

    // V stored transposed: Vt[(bh*HD + d)*SEQ + s]
    const size_t vtb = ((size_t)(b * NHEAD + h)) * HD * SEQ;
#pragma unroll
    for (int f = 0; f < 4; ++f)
      Vt[vtb + (size_t)(f * 16 + lm) * SEQ + sidx] = (bf16)accV[f][vr];
  }
}

// ---------------- kernel 2: causal flash attention ----------------
// K tiles staged into double-buffered LDS via async global->LDS copies
// (ASYNCcnt), overlapping the next chunk's copy with this chunk's WMMAs.
__global__ void __launch_bounds__(32)
attn_kernel(const bf16* __restrict__ Qb, const bf16* __restrict__ Kb,
            const bf16* __restrict__ Vt, bf16* __restrict__ ctx) {
  __shared__ __align__(16) bf16 lds_k[2][32 * HD];        // 2 x 4 KiB K stage
  __shared__ __align__(16) bf16 lds_p[16 * 32];           // P tile, 1 KiB
  const int lane = threadIdx.x & 31;
  const int lm = lane & 15, kg = lane >> 4;
  const int qtile = blockIdx.x;                           // 0..127
  const int bh    = blockIdx.y;                           // 0..31
  const int qbase = qtile * 16;

  const bf16* Qp = Qb + (size_t)bh * SEQ * HD;
  const bf16* Kp = Kb + (size_t)bh * SEQ * HD;
  const bf16* Vp = Vt + (size_t)bh * HD * SEQ;

  v16bf aQ[2];
#pragma unroll
  for (int hh = 0; hh < 2; ++hh) aQ[hh] = load_a_frag(Qp, HD, qbase, hh * 32, lane);

  v8f o[4] = {};
  float mrow[8], lrow[8];
#pragma unroll
  for (int vr = 0; vr < 8; ++vr) { mrow[vr] = -1.0e30f; lrow[vr] = 0.0f; }

  const int nchunks = (qbase >> 5) + 1;                   // causal: keys <= qbase+15

  // prologue: start async copy of chunk 0's K tile
  async_copy_chunk(Kp, lds_k[0], lane);

  for (int c = 0; c < nchunks; ++c) {
    const int kb = c * 32;
    const bf16* kl = lds_k[c & 1];

    // kick off next chunk's K copy, then wait for the current buffer
    // (async loads complete in order: <=8 outstanding => buffer c resident)
    if (c + 1 < nchunks) {
      async_copy_chunk(Kp + (size_t)(c + 1) * 32 * HD, lds_k[(c + 1) & 1], lane);
      asm volatile("s_wait_asynccnt 0x8" ::: "memory");
    } else {
      asm volatile("s_wait_asynccnt 0x0" ::: "memory");
    }

    // issue V fragments (global) and K fragments (LDS stage) up front; fence
    // so V loads stay in flight through S + softmax
    v16bf bV[4], bKf[2][2];
#pragma unroll
    for (int f = 0; f < 4; ++f) bV[f] = load_b_frag(Vp, SEQ, f * 16, kb, lane);
#pragma unroll
    for (int kblk = 0; kblk < 2; ++kblk)
#pragma unroll
      for (int hh = 0; hh < 2; ++hh)
        bKf[kblk][hh] = load_b_frag_lds(kl, kblk * 16, hh * 32, lane);
    SCHED_FENCE();

    // S = Q K^T for two 16-key blocks (K-dim = hd = 64 -> 2 steps each)
    v8f sfrag[2] = {};
#pragma unroll
    for (int kblk = 0; kblk < 2; ++kblk)
#pragma unroll
      for (int hh = 0; hh < 2; ++hh)
        sfrag[kblk] = WMMA_BF16(aQ[hh], bKf[kblk][hh], sfrag[kblk]);

    // causal mask + online softmax (row stats per D-fragment row)
#pragma unroll
    for (int vr = 0; vr < 8; ++vr) {
      const int qrow = qbase + vr + 8 * kg;
      float s0 = sfrag[0][vr]; if (kb + lm      > qrow) s0 = -1.0e30f;
      float s1 = sfrag[1][vr]; if (kb + 16 + lm > qrow) s1 = -1.0e30f;
      float mx = fmaxf(s0, s1);
#pragma unroll
      for (int msk = 1; msk < 16; msk <<= 1) mx = fmaxf(mx, __shfl_xor(mx, msk, 32));
      const float mnew  = fmaxf(mrow[vr], mx);
      const float alpha = __expf(mrow[vr] - mnew);
      mrow[vr] = mnew;
      const float p0 = __expf(s0 - mnew);
      const float p1 = __expf(s1 - mnew);
      float ps = p0 + p1;
#pragma unroll
      for (int msk = 1; msk < 16; msk <<= 1) ps += __shfl_xor(ps, msk, 32);
      lrow[vr] = lrow[vr] * alpha + ps;
      const int M = vr + 8 * kg;
      lds_p[M * 32 + lm]      = (bf16)p0;
      lds_p[M * 32 + 16 + lm] = (bf16)p1;
#pragma unroll
      for (int f = 0; f < 4; ++f) o[f][vr] *= alpha;
    }

    asm volatile("s_wait_dscnt 0x0" ::: "memory");        // LDS write -> read (same wave)

    // re-shape P (16x32) into an A fragment from LDS
    v16bf aP;
#pragma unroll
    for (int i = 0; i < 16; ++i) aP[i] = lds_p[lm * 32 + kg * 16 + i];

    // O += P @ V   (K-dim = 32 keys; B from transposed V, already in flight)
#pragma unroll
    for (int f = 0; f < 4; ++f) o[f] = WMMA_BF16(aP, bV[f], o[f]);
  }

  // normalize and merge heads: ctx[b][s][h*64 + d]
  const int b = bh >> 4, h = bh & 15;
#pragma unroll
  for (int vr = 0; vr < 8; ++vr) {
    const int row = qbase + vr + 8 * kg;
    const float inv_l = 1.0f / lrow[vr];
    const size_t base = ((size_t)b * SEQ + row) * DMODEL + h * HD;
#pragma unroll
    for (int f = 0; f < 4; ++f)
      ctx[base + f * 16 + lm] = (bf16)(o[f][vr] * inv_l);
  }
}

// ---------------- kernel 3: output projection (fp32 out) ----------------
__global__ void __launch_bounds__(32)
out_proj_kernel(const bf16* __restrict__ ctxb, const bf16* __restrict__ Wot,
                float* __restrict__ out) {
  const int lane = threadIdx.x & 31;
  const int lm = lane & 15, kg = lane >> 4;
  const int row_base = blockIdx.x * 16;
  const int nb = blockIdx.y * 64;

  v8f acc[4] = {};
#pragma unroll 2
  for (int kb = 0; kb < DMODEL; kb += 32) {
    __builtin_prefetch(ctxb + (size_t)(row_base + lm) * DMODEL + kb + 64, 0, 1);
    const v16bf a = load_a_frag(ctxb, DMODEL, row_base, kb, lane);
    v16bf bw[4];
#pragma unroll
    for (int f = 0; f < 4; ++f) bw[f] = load_b_frag(Wot, DMODEL, nb + f * 16, kb, lane);
    SCHED_FENCE();
#pragma unroll
    for (int f = 0; f < 4; ++f) acc[f] = WMMA_BF16(a, bw[f], acc[f]);
  }
#pragma unroll
  for (int vr = 0; vr < 8; ++vr) {
    const int row = row_base + vr + 8 * kg;
#pragma unroll
    for (int f = 0; f < 4; ++f)
      out[(size_t)row * DMODEL + nb + f * 16 + lm] = acc[f][vr];
  }
}

// ---------------- launcher ----------------
extern "C" void kernel_launch(void* const* d_in, const int* in_sizes, int n_in,
                              void* d_out, int out_size, void* d_ws, size_t ws_size,
                              hipStream_t stream) {
  const float* x  = (const float*)d_in[0];
  const float* Wq = (const float*)d_in[1];
  const float* Wk = (const float*)d_in[2];
  const float* Wv = (const float*)d_in[3];
  const float* Wo = (const float*)d_in[4];
  float* out = (float*)d_out;

  const size_t SZ_X   = (size_t)MROWS * DMODEL;   // 4 Mi elems
  const size_t SZ_W   = (size_t)DMODEL * DMODEL;  // 1 Mi elems
  const size_t SZ_QKV = (size_t)BATCH * NHEAD * SEQ * HD;

  bf16* xb   = (bf16*)d_ws;
  bf16* Wqt  = xb  + SZ_X;
  bf16* Wkt  = Wqt + SZ_W;
  bf16* Wvt  = Wkt + SZ_W;
  bf16* Wot  = Wvt + SZ_W;
  bf16* Qb   = Wot + SZ_W;
  bf16* Kb   = Qb  + SZ_QKV;
  bf16* Vt   = Kb  + SZ_QKV;
  bf16* ctxb = Vt  + SZ_QKV;

  // 0) precision conversion / layout
  cast_bf16_kernel<<<(int)(SZ_X / 256), 256, 0, stream>>>(x, xb, (int)SZ_X);
  transpose_cast_kernel<<<(int)(SZ_W / 256), 256, 0, stream>>>(Wq, Wqt);
  transpose_cast_kernel<<<(int)(SZ_W / 256), 256, 0, stream>>>(Wk, Wkt);
  transpose_cast_kernel<<<(int)(SZ_W / 256), 256, 0, stream>>>(Wv, Wvt);
  transpose_cast_kernel<<<(int)(SZ_W / 256), 256, 0, stream>>>(Wo, Wot);

  // 1) fused QKV projection + RoPE
  qkv_rope_kernel<<<dim3(MROWS / 16, NHEAD), 32, 0, stream>>>(xb, Wqt, Wkt, Wvt, Qb, Kb, Vt);

  // 2) causal flash attention (async K staging)
  attn_kernel<<<dim3(SEQ / 16, BATCH * NHEAD), 32, 0, stream>>>(Qb, Kb, Vt, ctxb);

  // 3) output projection
  out_proj_kernel<<<dim3(MROWS / 16, DMODEL / 64), 32, 0, stream>>>(ctxb, Wot, out);
}
